// EncoderLayer_80290118631980
// MI455X (gfx1250) — matrix operand order repaired and checked by reading, not compiled, not run on hardware
//
#include <hip/hip_runtime.h>

typedef __attribute__((ext_vector_type(16))) _Float16 v16h;
typedef __attribute__((ext_vector_type(8)))  _Float16 v8h;
typedef __attribute__((ext_vector_type(4)))  _Float16 v4h;
typedef __attribute__((ext_vector_type(8)))  float    v8f;

#define TBM 128
#define TBN 128
#define TBK 32
#define ROWP 40   // LDS row pitch in halves (80 B -> 16-B aligned fragment chunks)

union Frag16 { v16h v; v8h h[2]; };

// ---------------------------------------------------------------------------
// Tiled WMMA GEMM:  C = alpha * (A @ opB(B)) + bias + residual, opt. ReLU
// A: [M,K] f32 (lda); B: [K,N] f32 (ldb) or [N,K] if TRANSB; C: [M,N] f32.
// Batched via blockIdx.z (strides sA/sB/sC). f32->f16 conversion during LDS
// staging; f32 accumulation via v_wmma_f32_16x16x32_f16.
// Assumes M % 128 == 0 and K % 32 == 0 (true for every call in this layer);
// NGUARD enables N-dimension guards (only the context GEMM, N=64, needs it).
// Register-prefetch pipeline: next tile's global loads issue before this
// tile's fragment ds_loads + WMMAs; L2 prefetch runs one more tile ahead.
// ---------------------------------------------------------------------------
template<int TRANSB, int NGUARD>
__global__ __launch_bounds__(256) void gemm_wmma_kernel(
    const float* __restrict__ A, int lda, long long sA,
    const float* __restrict__ B, int ldb, long long sB,
    const float* __restrict__ bias,
    const float* __restrict__ residual,
    float* __restrict__ C, int ldc, long long sC,
    int M, int N, int K, float alpha, int relu)
{
    __shared__ _Float16 As[TBM * ROWP];
    __shared__ _Float16 Bs[TBN * ROWP];

    const int tid  = threadIdx.x;
    const int lane = tid & 31;
    const int wave = tid >> 5;
    const int wr   = wave & 3;    // 4 row groups of 32 rows
    const int wc   = wave >> 2;   // 2 col groups of 64 cols
    const int lm   = lane & 15;
    const int lh   = lane >> 4;

    A += (long long)blockIdx.z * sA;
    B += (long long)blockIdx.z * sB;
    C += (long long)blockIdx.z * sC;
    const int m0 = blockIdx.y * TBM;
    const int n0 = blockIdx.x * TBN;

    // Fixed per-thread staging coordinates.
    const int ar  = (tid * 4) / TBK;     // A (or Bt) row within tile (it adds 32)
    const int ac  = (tid * 4) % TBK;     // A (or Bt) col (k) within tile
    const int bn4 = (tid & 31) * 4;      // B: 4-col (n) block base (TRANSB=0)
    const int bk4 = (tid >> 5) * 4;      // B: 4-row (k) block base (TRANSB=0)

    v8f acc[2][4];
    #pragma unroll
    for (int i = 0; i < 2; ++i)
        #pragma unroll
        for (int j = 0; j < 4; ++j)
            #pragma unroll
            for (int v = 0; v < 8; ++v)
                acc[i][j][v] = 0.0f;

    float4 ra[4], rb[4];
    const bool bok = !NGUARD || (n0 + (TRANSB ? 0 : bn4)) < N; // uniform per thread

    auto load_tiles = [&](int k0) {
        #pragma unroll
        for (int it = 0; it < 4; ++it)
            ra[it] = *(const float4*)(A + (long long)(m0 + ar + it * 32) * lda + (k0 + ac));
        if (TRANSB) {
            #pragma unroll
            for (int it = 0; it < 4; ++it) {
                const int gn = n0 + ar + it * 32;
                if (!NGUARD || gn < N)
                    rb[it] = *(const float4*)(B + (long long)gn * ldb + (k0 + ac));
                else
                    rb[it] = make_float4(0.f, 0.f, 0.f, 0.f);
            }
        } else {
            #pragma unroll
            for (int u = 0; u < 4; ++u) {   // u: k row within 4x4 block
                if (bok)
                    rb[u] = *(const float4*)(B + (long long)(k0 + bk4 + u) * ldb + (n0 + bn4));
                else
                    rb[u] = make_float4(0.f, 0.f, 0.f, 0.f);
            }
        }
    };

    auto prefetch_tiles = [&](int k0) {
        __builtin_prefetch(A + (long long)(m0 + ar) * lda + (k0 + ac), 0, 3);
        if (TRANSB)
            __builtin_prefetch(B + (long long)(n0 + ar) * ldb + (k0 + ac), 0, 3);
        else
            __builtin_prefetch(B + (long long)(k0 + bk4) * ldb + (n0 + bn4), 0, 3);
    };

    auto stage_tiles = [&]() {
        #pragma unroll
        for (int it = 0; it < 4; ++it) {
            v4h hv;
            hv[0] = (_Float16)ra[it].x; hv[1] = (_Float16)ra[it].y;
            hv[2] = (_Float16)ra[it].z; hv[3] = (_Float16)ra[it].w;
            *(v4h*)&As[(ar + it * 32) * ROWP + ac] = hv;
        }
        if (TRANSB) {
            #pragma unroll
            for (int it = 0; it < 4; ++it) {
                v4h hv;
                hv[0] = (_Float16)rb[it].x; hv[1] = (_Float16)rb[it].y;
                hv[2] = (_Float16)rb[it].z; hv[3] = (_Float16)rb[it].w;
                *(v4h*)&Bs[(ar + it * 32) * ROWP + ac] = hv;
            }
        } else {
            // 4x4 in-register transpose: rows of rb are k, components are n.
            const float* rbf = (const float*)rb;   // rbf[k*4 + n]
            #pragma unroll
            for (int u = 0; u < 4; ++u) {          // u: n within block
                v4h hv;
                hv[0] = (_Float16)rbf[0 * 4 + u];
                hv[1] = (_Float16)rbf[1 * 4 + u];
                hv[2] = (_Float16)rbf[2 * 4 + u];
                hv[3] = (_Float16)rbf[3 * 4 + u];
                *(v4h*)&Bs[(bn4 + u) * ROWP + bk4] = hv;
            }
        }
    };

    const int ktiles = K / TBK;
    load_tiles(0);
    if (ktiles > 1) prefetch_tiles(TBK);

    for (int kt = 0; kt < ktiles; ++kt) {
        stage_tiles();                 // waits on global loads, converts, ds_store
        __syncthreads();

        if (kt + 1 < ktiles) {
            load_tiles((kt + 1) * TBK);     // register prefetch under the WMMAs
            if (kt + 2 < ktiles)
                prefetch_tiles((kt + 2) * TBK);  // L2 prefetch one tile further
        }

        // ---- fragments per ISA 7.12.2 layouts ----
        Frag16 af[2], bf[4];
        #pragma unroll
        for (int i = 0; i < 2; ++i) {
            const _Float16* p = &As[(wr * 32 + i * 16 + lm) * ROWP + lh * 8];
            af[i].h[0] = *(const v8h*)p;          // K = lh*8 .. +7
            af[i].h[1] = *(const v8h*)(p + 16);   // K = 16 + lh*8 ..
        }
        #pragma unroll
        for (int j = 0; j < 4; ++j) {
            const _Float16* p = &Bs[(wc * 64 + j * 16 + lm) * ROWP + lh * 8];
            bf[j].h[0] = *(const v8h*)p;          // K = lh*8 .. +7
            bf[j].h[1] = *(const v8h*)(p + 16);   // K = 16 + lh*8 ..
        }
        #pragma unroll
        for (int i = 0; i < 2; ++i)
            #pragma unroll
            for (int j = 0; j < 4; ++j)
                acc[i][j] = __builtin_amdgcn_wmma_f32_16x16x32_f16(
                    false, af[i].v, false, bf[j].v,
                    (short)0, acc[i][j], false, false);

        __syncthreads();
    }

    // ---- epilogue: alpha, bias, residual, relu, store ----
    #pragma unroll
    for (int i = 0; i < 2; ++i) {
        #pragma unroll
        for (int j = 0; j < 4; ++j) {
            const int col = n0 + wc * 64 + j * 16 + lm;
            if (NGUARD && col >= N) continue;
            const float bv = bias ? bias[col] : 0.0f;
            #pragma unroll
            for (int v = 0; v < 8; ++v) {
                const int row = m0 + wr * 32 + i * 16 + 8 * lh + v;
                float r = acc[i][j][v] * alpha + bv;
                if (residual) r += residual[(long long)row * ldc + col];
                if (relu) r = fmaxf(r, 0.0f);
                C[(long long)row * ldc + col] = r;
            }
        }
    }
}

// ---------------------------------------------------------------------------
// Row softmax over S=4096, one 256-thread block per row, in place, b128 I/O.
// ---------------------------------------------------------------------------
#define SEQ 4096
__global__ __launch_bounds__(256) void softmax_kernel(float* __restrict__ attn)
{
    float* p = attn + (long long)blockIdx.x * SEQ;
    const int tid = threadIdx.x, lane = tid & 31, wave = tid >> 5;
    const int off = tid * 4;
    __shared__ float red[8];
    __shared__ float bcast;

    union { float4 v4[4]; float f[16]; } e;
    #pragma unroll
    for (int i = 0; i < 4; ++i) e.v4[i] = *(const float4*)(p + i * 1024 + off);

    float m = -3.4e38f;
    #pragma unroll
    for (int i = 0; i < 16; ++i) m = fmaxf(m, e.f[i]);
    #pragma unroll
    for (int o = 16; o > 0; o >>= 1) m = fmaxf(m, __shfl_xor(m, o, 32));
    if (lane == 0) red[wave] = m;
    __syncthreads();
    if (tid == 0) {
        float t = red[0];
        for (int i = 1; i < 8; ++i) t = fmaxf(t, red[i]);
        bcast = t;
    }
    __syncthreads();
    m = bcast;

    float s = 0.f;
    #pragma unroll
    for (int i = 0; i < 16; ++i) { e.f[i] = __expf(e.f[i] - m); s += e.f[i]; }
    #pragma unroll
    for (int o = 16; o > 0; o >>= 1) s += __shfl_xor(s, o, 32);
    if (lane == 0) red[wave] = s;
    __syncthreads();
    if (tid == 0) {
        float t = 0.f;
        for (int i = 0; i < 8; ++i) t += red[i];
        bcast = 1.0f / t;
    }
    __syncthreads();
    const float inv = bcast;
    #pragma unroll
    for (int i = 0; i < 16; ++i) e.f[i] *= inv;
    #pragma unroll
    for (int i = 0; i < 4; ++i) *(float4*)(p + i * 1024 + off) = e.v4[i];
}

// ---------------------------------------------------------------------------
// LayerNorm over rows of 512, one 256-thread block per row, b64 I/O.
// ---------------------------------------------------------------------------
__global__ __launch_bounds__(256) void layernorm_kernel(
    const float* __restrict__ in, const float* __restrict__ gamma,
    const float* __restrict__ beta, float* __restrict__ out)
{
    const int tid = threadIdx.x, lane = tid & 31, wave = tid >> 5;
    const long long base = (long long)blockIdx.x * 512;
    __shared__ float red[8];
    __shared__ float bcast;

    const float2 v = *(const float2*)(in + base + tid * 2);
    float s = v.x + v.y;
    #pragma unroll
    for (int o = 16; o > 0; o >>= 1) s += __shfl_xor(s, o, 32);
    if (lane == 0) red[wave] = s;
    __syncthreads();
    if (tid == 0) {
        float t = 0.f;
        for (int i = 0; i < 8; ++i) t += red[i];
        bcast = t * (1.0f / 512.0f);
    }
    __syncthreads();
    const float mu = bcast;

    const float d0 = v.x - mu, d1 = v.y - mu;
    float q = d0 * d0 + d1 * d1;
    #pragma unroll
    for (int o = 16; o > 0; o >>= 1) q += __shfl_xor(q, o, 32);
    if (lane == 0) red[wave] = q;
    __syncthreads();
    if (tid == 0) {
        float t = 0.f;
        for (int i = 0; i < 8; ++i) t += red[i];
        bcast = t * (1.0f / 512.0f);
    }
    __syncthreads();
    const float inv = rsqrtf(bcast + 1e-5f);

    const float2 g = *(const float2*)(gamma + tid * 2);
    const float2 b = *(const float2*)(beta + tid * 2);
    float2 o2;
    o2.x = d0 * inv * g.x + b.x;
    o2.y = d1 * inv * g.y + b.y;
    *(float2*)(out + base + tid * 2) = o2;
}

// ---------------------------------------------------------------------------
extern "C" void kernel_launch(void* const* d_in, const int* in_sizes, int n_in,
                              void* d_out, int out_size, void* d_ws, size_t ws_size,
                              hipStream_t stream)
{
    const int S = 4096, D = 512, F = 2048, H = 8, DPH = 64;
    const float* x  = (const float*)d_in[0];
    const float* wq = (const float*)d_in[1];
    const float* bq = (const float*)d_in[2];
    const float* wk = (const float*)d_in[3];
    const float* bk = (const float*)d_in[4];
    const float* wv = (const float*)d_in[5];
    const float* bv = (const float*)d_in[6];
    const float* wo = (const float*)d_in[7];
    const float* bo = (const float*)d_in[8];
    const float* w1 = (const float*)d_in[9];
    const float* b1 = (const float*)d_in[10];
    const float* w2 = (const float*)d_in[11];
    const float* b2 = (const float*)d_in[12];
    const float* gamma = (const float*)d_in[13];
    const float* beta  = (const float*)d_in[14];

    float* out  = (float*)d_out;                 // [S, D]
    float* attn = out + (size_t)S * D;           // [H, S, S] (written here directly)

    float* ws = (float*)d_ws;
    const size_t SD = (size_t)S * D;
    float* Q        = ws;              // [S, D]
    float* Kb       = Q + SD;          // [S, D]
    float* V        = Kb + SD;         // [S, D]
    float* ctx      = V + SD;          // [S, D]
    float* attn_out = ctx + SD;        // [S, D]
    float* ffn_sum  = attn_out + SD;   // [S, D]
    float* ffn1     = ffn_sum + SD;    // [S, F]

    const dim3 blk(256);
    const long long hs = (long long)S * DPH;     // per-head stride in Q/K/V/ctx
    const long long ss = (long long)S * S;       // per-head stride in attn

    // 1) QKV projections: y = x @ W + b
    gemm_wmma_kernel<0,0><<<dim3(D/TBN, S/TBM, 1), blk, 0, stream>>>(
        x, D, 0, wq, D, 0, bq, nullptr, Q,  D, 0, S, D, D, 1.f, 0);
    gemm_wmma_kernel<0,0><<<dim3(D/TBN, S/TBM, 1), blk, 0, stream>>>(
        x, D, 0, wk, D, 0, bk, nullptr, Kb, D, 0, S, D, D, 1.f, 0);
    gemm_wmma_kernel<0,0><<<dim3(D/TBN, S/TBM, 1), blk, 0, stream>>>(
        x, D, 0, wv, D, 0, bv, nullptr, V,  D, 0, S, D, D, 1.f, 0);

    // 2) scores = Q_h @ K_h^T / sqrt(64); heads via the flat reshape (stride S*DPH)
    gemm_wmma_kernel<1,0><<<dim3(S/TBN, S/TBM, H), blk, 0, stream>>>(
        Q, DPH, hs, Kb, DPH, hs, nullptr, nullptr,
        attn, S, ss, S, S, DPH, 0.125f, 0);

    // 3) softmax rows in place (result is the returned attn tensor)
    softmax_kernel<<<dim3(H * S), blk, 0, stream>>>(attn);

    // 4) context = attn_h @ V_h  (N=64 < TBN -> guarded instantiation)
    gemm_wmma_kernel<0,1><<<dim3(1, S/TBM, H), blk, 0, stream>>>(
        attn, S, ss, V, DPH, hs, nullptr, nullptr,
        ctx, DPH, hs, S, DPH, S, 1.f, 0);

    // 5) attn_out = ctx @ wo + bo + x
    gemm_wmma_kernel<0,0><<<dim3(D/TBN, S/TBM, 1), blk, 0, stream>>>(
        ctx, D, 0, wo, D, 0, bo, x, attn_out, D, 0, S, D, D, 1.f, 0);

    // 6) ffn1 = relu(attn_out @ w1 + b1)
    gemm_wmma_kernel<0,0><<<dim3(F/TBN, S/TBM, 1), blk, 0, stream>>>(
        attn_out, D, 0, w1, F, 0, b1, nullptr, ffn1, F, 0, S, F, D, 1.f, 1);

    // 7) ffn_sum = ffn1 @ w2 + b2 + attn_out
    gemm_wmma_kernel<0,0><<<dim3(D/TBN, S/TBM, 1), blk, 0, stream>>>(
        ffn1, F, 0, w2, D, 0, b2, attn_out, ffn_sum, D, 0, S, D, F, 1.f, 0);

    // 8) out = LayerNorm(ffn_sum) * gamma + beta
    layernorm_kernel<<<dim3(S), blk, 0, stream>>>(ffn_sum, gamma, beta, out);

    (void)in_sizes; (void)n_in; (void)out_size; (void)ws_size;
}
